// RobustSymGAT_42245298324094
// MI455X (gfx1250) — matrix operand order repaired and checked by reading, not compiled, not run on hardware
//
#include <hip/hip_runtime.h>

typedef __attribute__((ext_vector_type(16))) _Float16 v16h;
typedef __attribute__((ext_vector_type(8)))  float    v8f;

#define EPSF 1e-8f

// ---- LDS layout constants (element units) ----
// f16 transposed weights Wt[n][k], row stride 72 halves (144B -> bank-conflict-free)
#define WT_IN   0            // 64 x 72
#define WT_1    4608         // 64 x 72
#define WT_2    9216         // 64 x 72
#define WT_3    13824        // 64 x 72
#define WT_R1   18432        // 32 x 72
#define WT_C1   20736        // 32 x 72
#define WT_SYM  23040        // 32 x 72  (Wsym^T, K padded 10->32)
#define WT_TOT  25344
// f32 constants
#define C_BSYM  0            // 32
#define C_BIN   32           // 64
#define C_B1    96           // 64
#define C_B2    160          // 64
#define C_B3    224          // 64
#define C_BR1   288          // 32
#define C_BC1   320          // 32
#define C_WR2   352          // 32
#define C_BR2   384          // 1
#define C_WC2   385          // 32 x 4
#define C_BC2   513          // 4
#define C_TOT   520

union AFrag  { v16h v; uint4 q[2]; };
union Row32h { _Float16 h[32]; uint4 q[4]; };
union Pairh  { _Float16 h[2];  unsigned int u; };

__device__ __forceinline__ float elu_f(float x) {
    return x > 0.0f ? x : (__expf(x) - 1.0f);
}
__device__ __forceinline__ float pen_f(float v, float lo, float hi) {
    float below = (lo - v) / lo;
    float above = (v - hi) / hi;
    return v < lo ? below : (v > hi ? above : 0.0f);
}

__launch_bounds__(256, 2)
__global__ void fused_symgat_mlp(
    const float* __restrict__ x,
    const float* __restrict__ Wsym, const float* __restrict__ bsym,
    const float* __restrict__ Win,  const float* __restrict__ bin,
    const float* __restrict__ W1,   const float* __restrict__ b1,
    const float* __restrict__ W2,   const float* __restrict__ b2,
    const float* __restrict__ W3,   const float* __restrict__ b3,
    const float* __restrict__ Wr1,  const float* __restrict__ br1,
    const float* __restrict__ Wr2,  const float* __restrict__ br2,
    const float* __restrict__ Wc1,  const float* __restrict__ bc1,
    const float* __restrict__ Wc2,  const float* __restrict__ bc2,
    float* __restrict__ out, int nrows, int ntiles)
{
    __shared__ __align__(16) _Float16 sWT[WT_TOT];
    __shared__ __align__(16) _Float16 sAct[8 * 32 * 72];   // per-wave 32x72 f16 staging
    __shared__ __align__(16) float    sC[C_TOT];

    const int tid  = threadIdx.x;
    const int lane = tid & 31;
    const int wave = tid >> 5;
    const int hsel = lane >> 4;     // half-wave select (ISA fragment layout)
    const int lr   = lane & 15;     // row (A/M) or column (B/N, C/N) within a 16-tile

    // ---- stage weights to LDS: f32 -> f16, transposed, zero-padded ----
    for (int i = tid; i < 64 * 64; i += 256) {
        int k = i >> 6, n = i & 63;
        sWT[WT_IN + n * 72 + k] = (_Float16)((k < 42) ? Win[k * 64 + n] : 0.0f);
        sWT[WT_1  + n * 72 + k] = (_Float16)W1[i];
        sWT[WT_2  + n * 72 + k] = (_Float16)W2[i];
        sWT[WT_3  + n * 72 + k] = (_Float16)W3[i];
    }
    for (int i = tid; i < 64 * 32; i += 256) {
        int k = i >> 5, n = i & 31;
        sWT[WT_R1 + n * 72 + k] = (_Float16)Wr1[i];
        sWT[WT_C1 + n * 72 + k] = (_Float16)Wc1[i];
    }
    for (int i = tid; i < 32 * 32; i += 256) {
        int k = i >> 5, n = i & 31;
        sWT[WT_SYM + n * 72 + k] = (_Float16)((k < 10) ? Wsym[k * 32 + n] : 0.0f);
    }
    if (tid < 64) { sC[C_BIN + tid] = bin[tid]; sC[C_B1 + tid] = b1[tid];
                    sC[C_B2 + tid] = b2[tid];  sC[C_B3 + tid] = b3[tid]; }
    if (tid < 32) { sC[C_BSYM + tid] = bsym[tid]; sC[C_BR1 + tid] = br1[tid];
                    sC[C_BC1 + tid] = bc1[tid];  sC[C_WR2 + tid] = Wr2[tid]; }
    for (int i = tid; i < 128; i += 256) sC[C_WC2 + i] = Wc2[i];
    if (tid == 0) sC[C_BR2] = br2[0];
    if (tid < 4)  sC[C_BC2 + tid] = bc2[tid];
    __syncthreads();

    _Float16* act = &sAct[wave * 32 * 72];   // private to this wave: no block barriers needed

    // A fragment (16x32 f16) from row-major activation staging.
    // lanes 0-15: M=lr, K = ks*32 + {0..7, 16..23}; lanes 16-31: K = +8.
    auto loadA = [&](int mt, int ks) -> v16h {
        AFrag f;
        const _Float16* p = act + (mt * 16 + lr) * 72 + ks * 32 + hsel * 8;
        f.q[0] = *(const uint4*)p;          // K base .. base+7
        f.q[1] = *(const uint4*)(p + 16);   // K base+16 .. base+23
        return f.v;
    };
    // B fragment (32x16 f16) from transposed weights Wt[n][k].
    // lanes 0-15: N=lr, K = ks*32 + {0..7, 16..23}; lanes 16-31: K = +8.
    auto loadB = [&](int wtOff, int nt, int ks) -> v16h {
        AFrag f;
        const _Float16* p = &sWT[wtOff + (nt * 16 + lr) * 72 + ks * 32 + hsel * 8];
        f.q[0] = *(const uint4*)p;          // K base .. base+7
        f.q[1] = *(const uint4*)(p + 16);   // K base+16 .. base+23
        return f.v;
    };

    float h[2][4][8];   // persistent activations in C-layout: [mtile][ntile][vgpr]

    // dense 64->64 (+ optional residual), f16 WMMA with f32 accumulate
    auto dense64 = [&](int wtOff, int biasOff, bool resid) {
        #pragma unroll
        for (int nt = 0; nt < 4; ++nt) {
            float bn = sC[biasOff + nt * 16 + lr];   // C-layout: N = nt*16 + lr for all 8 VGPRs
            v8f a0 = {0,0,0,0,0,0,0,0};
            v8f a1 = {0,0,0,0,0,0,0,0};
            #pragma unroll
            for (int ks = 0; ks < 2; ++ks) {
                v16h bm = loadB(wtOff, nt, ks);
                v16h am0 = loadA(0, ks);
                a0 = __builtin_amdgcn_wmma_f32_16x16x32_f16(false, am0, false, bm,
                                                            (short)0, a0, false, false);
                v16h am1 = loadA(1, ks);
                a1 = __builtin_amdgcn_wmma_f32_16x16x32_f16(false, am1, false, bm,
                                                            (short)0, a1, false, false);
            }
            #pragma unroll
            for (int v = 0; v < 8; ++v) {
                float e0 = elu_f(a0[v] + bn);
                float e1 = elu_f(a1[v] + bn);
                h[0][nt][v] = resid ? h[0][nt][v] + e0 : e0;
                h[1][nt][v] = resid ? h[1][nt][v] + e1 : e1;
            }
        }
        // flush h -> f16 staging (C-layout: row = v + 8*hsel, col = nt*16 + lr)
        #pragma unroll
        for (int mt = 0; mt < 2; ++mt)
            #pragma unroll
            for (int nt = 0; nt < 4; ++nt)
                #pragma unroll
                for (int v = 0; v < 8; ++v)
                    act[(mt * 16 + v + 8 * hsel) * 72 + nt * 16 + lr] = (_Float16)h[mt][nt][v];
        __builtin_amdgcn_wave_barrier();   // keep flush before next layer's A loads
    };

    // head: 64->32 WMMA + ELU, results kept in registers (C-layout)
    auto head32 = [&](int wtOff, int biasOff, float e[2][2][8]) {
        #pragma unroll
        for (int nt = 0; nt < 2; ++nt) {
            float bn = sC[biasOff + nt * 16 + lr];
            v8f a0 = {0,0,0,0,0,0,0,0};
            v8f a1 = {0,0,0,0,0,0,0,0};
            #pragma unroll
            for (int ks = 0; ks < 2; ++ks) {
                v16h bm = loadB(wtOff, nt, ks);
                v16h am0 = loadA(0, ks);
                a0 = __builtin_amdgcn_wmma_f32_16x16x32_f16(false, am0, false, bm,
                                                            (short)0, a0, false, false);
                v16h am1 = loadA(1, ks);
                a1 = __builtin_amdgcn_wmma_f32_16x16x32_f16(false, am1, false, bm,
                                                            (short)0, a1, false, false);
            }
            #pragma unroll
            for (int v = 0; v < 8; ++v) {
                e[0][nt][v] = elu_f(a0[v] + bn);
                e[1][nt][v] = elu_f(a1[v] + bn);
            }
        }
    };

    for (int tile = blockIdx.x; tile < ntiles; tile += gridDim.x) {
        const int rowBase = tile * 256 + wave * 32;
        const int rg = rowBase + lane;          // this lane's row for elementwise work

        // speculative prefetch of next tile's x
        size_t pfrow = (size_t)rg + (size_t)gridDim.x * 256;
        if (pfrow < (size_t)nrows) __builtin_prefetch(x + pfrow * 10, 0, 1);

        // ---- stage 1: load x, symbolic penalty features (f32 VALU) ----
        float xv[10];
        const float2* xp = (const float2*)(x + (size_t)rg * 10);   // 8B aligned (stride 40B)
        #pragma unroll
        for (int i = 0; i < 5; ++i) { float2 t = xp[i]; xv[2 * i] = t.x; xv[2 * i + 1] = t.y; }

        float am = xv[0], bod = xv[1], dox = xv[2], ph = xv[4], nit = xv[7];
        float p_ph  = pen_f(ph,  6.5f,   8.5f);
        float p_am  = pen_f(am,  0.001f, 0.5f);
        float p_bod = pen_f(bod, 0.001f, 5.0f);
        float p_do  = dox < 6.0f ? (6.0f - dox) / 6.0f : 0.0f;      // hi = inf
        float p_nit = pen_f(nit, 0.001f, 10.0f);
        float sym[10];
        sym[0] = p_ph; sym[1] = p_am; sym[2] = p_bod; sym[3] = p_do; sym[4] = p_nit;
        sym[5] = (1.2f * am / (0.5f + EPSF) + 1.5f * bod / (5.0f + EPSF)
                  - 0.8f * dox / (10.0f + EPSF)) / 3.0f;
        sym[6] = (1.5f * ph / (8.5f + EPSF) + nit / (10.0f + EPSF)) / 2.0f;
        sym[7] = (2.0f * bod / (5.0f + EPSF) - 1.5f * dox / (10.0f + EPSF)
                  + 0.8f * am / (0.5f + EPSF)) / 3.0f;
        sym[8] = 2.0f * nit / (10.0f + EPSF);
        sym[9] = 1.0f / (1.0f + (p_ph + p_am + p_bod + p_do + p_nit) + EPSF);

        // write sym row (cols 0..31, K zero-padded) row-per-lane, 4x ds_store_b128
        _Float16* myrow = act + lane * 72;
        {
            Row32h r;
            #pragma unroll
            for (int i = 0; i < 10; ++i) r.h[i] = (_Float16)sym[i];
            #pragma unroll
            for (int i = 10; i < 32; ++i) r.h[i] = (_Float16)0.0f;
            #pragma unroll
            for (int qi = 0; qi < 4; ++qi) *((uint4*)myrow + qi) = r.q[qi];
        }
        __builtin_amdgcn_wave_barrier();

        // ---- sym encoder on the matrix pipe: elu(sym @ Wsym + bsym), K=32, N=32 ----
        float es[2][2][8];
        #pragma unroll
        for (int nt = 0; nt < 2; ++nt) {
            float bn = sC[C_BSYM + nt * 16 + lr];
            v8f a0 = {0,0,0,0,0,0,0,0};
            v8f a1 = {0,0,0,0,0,0,0,0};
            v16h bm = loadB(WT_SYM, nt, 0);
            v16h am0 = loadA(0, 0);
            a0 = __builtin_amdgcn_wmma_f32_16x16x32_f16(false, am0, false, bm,
                                                        (short)0, a0, false, false);
            v16h am1 = loadA(1, 0);
            a1 = __builtin_amdgcn_wmma_f32_16x16x32_f16(false, am1, false, bm,
                                                        (short)0, a1, false, false);
            #pragma unroll
            for (int v = 0; v < 8; ++v) {
                es[0][nt][v] = elu_f(a0[v] + bn);
                es[1][nt][v] = elu_f(a1[v] + bn);
            }
        }
        __builtin_amdgcn_wave_barrier();

        // ---- assemble padded input row [x(10) | sym_enc(32) | 0(22)] ----
        {   // x part: cols 0..9 (row-per-lane), one b128 + one b32
            Row32h r;
            #pragma unroll
            for (int i = 0; i < 8; ++i) r.h[i] = (_Float16)xv[i];
            *(uint4*)myrow = r.q[0];
            Pairh p2; p2.h[0] = (_Float16)xv[8]; p2.h[1] = (_Float16)xv[9];
            *(unsigned int*)(myrow + 8) = p2.u;
            // zero cols 32..63 (cols 32..41 get overwritten by sym_enc flush below)
            uint4 z; z.x = z.y = z.z = z.w = 0u;
            #pragma unroll
            for (int qi = 0; qi < 4; ++qi) *((uint4*)(myrow + 32) + qi) = z;
        }
        // sym_enc: cols 10..41, C-layout scatter
        #pragma unroll
        for (int mt = 0; mt < 2; ++mt)
            #pragma unroll
            for (int nt = 0; nt < 2; ++nt)
                #pragma unroll
                for (int v = 0; v < 8; ++v)
                    act[(mt * 16 + v + 8 * hsel) * 72 + 10 + nt * 16 + lr] = (_Float16)es[mt][nt][v];
        __builtin_amdgcn_wave_barrier();

        // ---- trunk ----
        dense64(WT_IN, C_BIN, false);
        dense64(WT_1,  C_B1,  true);
        dense64(WT_2,  C_B2,  true);
        dense64(WT_3,  C_B3,  true);

        // ---- regression head: elu(h@Wr1+br1) @ Wr2 + br2 ----
        {
            float er[2][2][8];
            head32(WT_R1, C_BR1, er);
            float w2a = sC[C_WR2 + lr], w2b = sC[C_WR2 + 16 + lr];
            #pragma unroll
            for (int mt = 0; mt < 2; ++mt) {
                float p[8];
                #pragma unroll
                for (int v = 0; v < 8; ++v)
                    p[v] = er[mt][0][v] * w2a + er[mt][1][v] * w2b;
                #pragma unroll
                for (int m = 1; m < 16; m <<= 1)
                    #pragma unroll
                    for (int v = 0; v < 8; ++v)
                        p[v] += __shfl_xor(p[v], m, 32);   // butterfly within 16-lane half
                if (lr == 0) {
                    float c = sC[C_BR2];
                    #pragma unroll
                    for (int v = 0; v < 8; ++v) {
                        int row = rowBase + mt * 16 + 8 * hsel + v;
                        out[row] = p[v] + c;
                    }
                }
            }
        }
        // ---- classification head: elu(h@Wc1+bc1) @ Wc2 + bc2 ----
        {
            float ec[2][2][8];
            head32(WT_C1, C_BC1, ec);
            float wc0[4], wc1[4];
            #pragma unroll
            for (int j = 0; j < 4; ++j) {
                wc0[j] = sC[C_WC2 + lr * 4 + j];
                wc1[j] = sC[C_WC2 + (16 + lr) * 4 + j];
            }
            #pragma unroll
            for (int mt = 0; mt < 2; ++mt) {
                float q[4][8];
                #pragma unroll
                for (int j = 0; j < 4; ++j)
                    #pragma unroll
                    for (int v = 0; v < 8; ++v)
                        q[j][v] = ec[mt][0][v] * wc0[j] + ec[mt][1][v] * wc1[j];
                #pragma unroll
                for (int m = 1; m < 16; m <<= 1)
                    #pragma unroll
                    for (int j = 0; j < 4; ++j)
                        #pragma unroll
                        for (int v = 0; v < 8; ++v)
                            q[j][v] += __shfl_xor(q[j][v], m, 32);
                if (lr == 0) {
                    #pragma unroll
                    for (int v = 0; v < 8; ++v) {
                        int row = rowBase + mt * 16 + 8 * hsel + v;
                        float4 o;
                        o.x = q[0][v] + sC[C_BC2 + 0];
                        o.y = q[1][v] + sC[C_BC2 + 1];
                        o.z = q[2][v] + sC[C_BC2 + 2];
                        o.w = q[3][v] + sC[C_BC2 + 3];
                        *(float4*)(out + (size_t)nrows + (size_t)row * 4) = o;
                    }
                }
            }
        }
    }
}

extern "C" void kernel_launch(void* const* d_in, const int* in_sizes, int n_in,
                              void* d_out, int out_size, void* d_ws, size_t ws_size,
                              hipStream_t stream) {
    (void)n_in; (void)out_size; (void)d_ws; (void)ws_size;
    const float* xin  = (const float*)d_in[0];
    const float* Wsym = (const float*)d_in[1];  const float* bsym = (const float*)d_in[2];
    const float* Win  = (const float*)d_in[3];  const float* bin  = (const float*)d_in[4];
    const float* W1   = (const float*)d_in[5];  const float* b1   = (const float*)d_in[6];
    const float* W2   = (const float*)d_in[7];  const float* b2   = (const float*)d_in[8];
    const float* W3   = (const float*)d_in[9];  const float* b3   = (const float*)d_in[10];
    const float* Wr1  = (const float*)d_in[11]; const float* br1  = (const float*)d_in[12];
    const float* Wr2  = (const float*)d_in[13]; const float* br2  = (const float*)d_in[14];
    const float* Wc1  = (const float*)d_in[15]; const float* bc1  = (const float*)d_in[16];
    const float* Wc2  = (const float*)d_in[17]; const float* bc2  = (const float*)d_in[18];

    int nrows  = in_sizes[0] / 10;          // B = 1,048,576
    int ntiles = nrows / 256;               // 256 rows per block-tile
    int grid   = ntiles < 2048 ? ntiles : 2048;   // persistent blocks, 2 tiles each at B=1M

    fused_symgat_mlp<<<dim3(grid), dim3(256), 0, stream>>>(
        xin, Wsym, bsym, Win, bin, W1, b1, W2, b2, W3, b3,
        Wr1, br1, Wr2, br2, Wc1, bc1, Wc2, bc2,
        (float*)d_out, nrows, ntiles);
}
